// SVDHead_23665269801354
// MI455X (gfx1250) — matrix operand order repaired and checked by reading, not compiled, not run on hardware
//
#include <hip/hip_runtime.h>

typedef __attribute__((ext_vector_type(16))) _Float16 v16h;
typedef __attribute__((ext_vector_type(8)))  float    v8f;

#define BATCH 64
#define NPTS  1024
#define MPTS  1024

union FragU {
    v16h v;
    uint4 q[2];
    _Float16 h[16];
};

// ---------------------------------------------------------------------------
// Kernel 1: fused softmax + (tgt,1) x p^T contraction via WMMA f16.
// Grid: 64 batches * 8 n-tiles-of-128 = 512 blocks, 256 threads (8 waves).
// Each wave handles 16 rows of scores (one WMMA N-tile).
// Writes corr[b][d][n] = sum_m tgt[d][m]*softmax(scores[n,:])[m]
// ---------------------------------------------------------------------------
__global__ __launch_bounds__(256)
void svdhead_corr_kernel(const float* __restrict__ tgt,
                         const float* __restrict__ scores,
                         float* __restrict__ corr) {
    // A-fragment image: 32 k-steps * 32 lanes * 2 uint4 = 32 KB
    __shared__ uint4 Ards[32 * 64];

    const int b    = blockIdx.x >> 3;
    const int tile = blockIdx.x & 7;
    const int tid  = threadIdx.x;
    const int lane = tid & 31;
    const int wave = tid >> 5;

    // ---- Build A fragments (rows: tgt_x, tgt_y, tgt_z, ones, 0..0) ----
    // 16-bit A 16x32 k order used here (contiguous; B uses the same):
//   lanes 0-15 : dwords0-7 = K0..15   (row M = lane)
//   lanes 16-31: dwords0-7 = K16..31  (row M = lane-16)
    const float* tb = tgt + (size_t)b * 3 * MPTS;
    for (int f = tid; f < 1024; f += 256) {
        const int s  = f >> 5;     // k-step, m_base = 32*s
        const int l  = f & 31;     // lane whose fragment we build
        const int r  = l & 15;     // matrix row (d)
        const int hi = l >> 4;
        const int m0 = s * 32 + hi * 16;
        FragU u;
        if (r < 3) {
            const float* trow = tb + r * MPTS;
#pragma unroll
            for (int j = 0; j < 8; ++j) {
                u.h[j]     = (_Float16)trow[m0 + j];
                u.h[8 + j] = (_Float16)trow[m0 + 8 + j];
            }
        } else if (r == 3) {
#pragma unroll
            for (int j = 0; j < 16; ++j) u.h[j] = (_Float16)1.0f;
        } else {
#pragma unroll
            for (int j = 0; j < 16; ++j) u.h[j] = (_Float16)0.0f;
        }
        Ards[s * 64 +      l] = u.q[0];
        Ards[s * 64 + 32 + l] = u.q[1];
    }
    __syncthreads();

    // ---- Per-wave 16-row tile ----
    const int nbase = tile * 128 + wave * 16;
    const int row   = nbase + (lane & 15);
    const int hi    = lane >> 4;
    const float* srow = scores + ((size_t)b * NPTS + row) * MPTS;

    // Pass 1: row max (each of the 2 lanes per row scans half the row)
    float mx = -3.402823466e38f;
    {
        const float4* sv = reinterpret_cast<const float4*>(srow) + hi * 128;
#pragma unroll 4
        for (int i = 0; i < 128; ++i) {
            float4 v = sv[i];
            mx = fmaxf(mx, fmaxf(fmaxf(v.x, v.y), fmaxf(v.z, v.w)));
        }
    }
    const float rowmax = fmaxf(mx, __shfl_xor(mx, 16, 32));

    // Pass 2: stream exp values into B fragments, accumulate with WMMA.
    // B 32x16 f16 layout: lane holds col N = lane&15; lanes 0-15 K=0..15,
    // lanes 16-31 K=16..31 (halves in ascending K).
    v8f acc = {};
#pragma unroll 2
    for (int s = 0; s < 32; ++s) {
        const int m0 = s * 32 + hi * 16;
        float4 a0 = *reinterpret_cast<const float4*>(srow + m0);
        float4 a1 = *reinterpret_cast<const float4*>(srow + m0 + 4);
        float4 a2 = *reinterpret_cast<const float4*>(srow + m0 + 8);
        float4 a3 = *reinterpret_cast<const float4*>(srow + m0 + 12);
        FragU bu;
        bu.h[0]  = (_Float16)__expf(a0.x - rowmax);
        bu.h[1]  = (_Float16)__expf(a0.y - rowmax);
        bu.h[2]  = (_Float16)__expf(a0.z - rowmax);
        bu.h[3]  = (_Float16)__expf(a0.w - rowmax);
        bu.h[4]  = (_Float16)__expf(a1.x - rowmax);
        bu.h[5]  = (_Float16)__expf(a1.y - rowmax);
        bu.h[6]  = (_Float16)__expf(a1.z - rowmax);
        bu.h[7]  = (_Float16)__expf(a1.w - rowmax);
        bu.h[8]  = (_Float16)__expf(a2.x - rowmax);
        bu.h[9]  = (_Float16)__expf(a2.y - rowmax);
        bu.h[10] = (_Float16)__expf(a2.z - rowmax);
        bu.h[11] = (_Float16)__expf(a2.w - rowmax);
        bu.h[12] = (_Float16)__expf(a3.x - rowmax);
        bu.h[13] = (_Float16)__expf(a3.y - rowmax);
        bu.h[14] = (_Float16)__expf(a3.z - rowmax);
        bu.h[15] = (_Float16)__expf(a3.w - rowmax);

        FragU au;
        au.q[0] = Ards[s * 64 + lane];
        au.q[1] = Ards[s * 64 + 32 + lane];

        acc = __builtin_amdgcn_wmma_f32_16x16x32_f16(
            /*neg_a=*/false, au.v, /*neg_b=*/false, bu.v,
            /*c_mod=*/(short)0, acc, /*reuse_a=*/false, /*reuse_b=*/false);
    }

    // D rows 0..3 live in VGPRs 0..3 of lanes 0..15 (N = lane)
    if (lane < 16) {
        const int n = nbase + lane;
        const float inv = 1.0f / acc[3];   // denom >= 1 (max element contributes 1)
        float* cb = corr + (size_t)b * 3 * NPTS;
        cb[0 * NPTS + n] = acc[0] * inv;
        cb[1 * NPTS + n] = acc[1] * inv;
        cb[2 * NPTS + n] = acc[2] * inv;
    }
}

// ---------------------------------------------------------------------------
// Kernel 2: per-batch reductions + 3x3 Jacobi SVD + Kabsch. 1 block / batch.
// ---------------------------------------------------------------------------
__global__ __launch_bounds__(256)
void svdhead_kabsch_kernel(const float* __restrict__ src,
                           const float* __restrict__ corr,
                           float* __restrict__ out) {
    __shared__ float red[256];
    __shared__ float tot[15];
    const int b   = blockIdx.x;
    const int tid = threadIdx.x;
    const float* sp = src  + (size_t)b * 3 * NPTS;
    const float* cp = corr + (size_t)b * 3 * NPTS;

    float p[15];
#pragma unroll
    for (int i = 0; i < 15; ++i) p[i] = 0.0f;
    for (int n = tid; n < NPTS; n += 256) {
        float s0 = sp[n], s1 = sp[NPTS + n], s2 = sp[2 * NPTS + n];
        float c0 = cp[n], c1 = cp[NPTS + n], c2 = cp[2 * NPTS + n];
        p[0] += s0; p[1] += s1; p[2] += s2;
        p[3] += c0; p[4] += c1; p[5] += c2;
        p[6]  += s0 * c0; p[7]  += s0 * c1; p[8]  += s0 * c2;
        p[9]  += s1 * c0; p[10] += s1 * c1; p[11] += s1 * c2;
        p[12] += s2 * c0; p[13] += s2 * c1; p[14] += s2 * c2;
    }
    for (int q = 0; q < 15; ++q) {
        red[tid] = p[q];
        __syncthreads();
        for (int off = 128; off > 0; off >>= 1) {
            if (tid < off) red[tid] += red[tid + off];
            __syncthreads();
        }
        if (tid == 0) tot[q] = red[0];
        __syncthreads();
    }

    if (tid == 0) {
        const float invN = 1.0f / (float)NPTS;
        float mu[3] = { tot[0] * invN, tot[1] * invN, tot[2] * invN };
        float nu[3] = { tot[3] * invN, tot[4] * invN, tot[5] * invN };
        float H[3][3];
        for (int d = 0; d < 3; ++d)
            for (int e = 0; e < 3; ++e)
                H[d][e] = tot[6 + d * 3 + e] - (float)NPTS * mu[d] * nu[e];

        // K = H^T H (symmetric), Jacobi eigen-decomposition -> V, eig
        float K[3][3];
        for (int i = 0; i < 3; ++i)
            for (int j = 0; j < 3; ++j)
                K[i][j] = H[0][i] * H[0][j] + H[1][i] * H[1][j] + H[2][i] * H[2][j];
        float V[3][3] = { {1,0,0},{0,1,0},{0,0,1} };
        for (int sweep = 0; sweep < 10; ++sweep) {
            for (int pair = 0; pair < 3; ++pair) {
                const int pp = (pair == 2) ? 1 : 0;
                const int qq = (pair == 0) ? 1 : 2;
                float apq = K[pp][qq];
                if (fabsf(apq) < 1e-30f) continue;
                float theta = (K[qq][qq] - K[pp][pp]) / (2.0f * apq);
                float tt = ((theta >= 0.0f) ? 1.0f : -1.0f) /
                           (fabsf(theta) + sqrtf(theta * theta + 1.0f));
                float cc = 1.0f / sqrtf(tt * tt + 1.0f);
                float ss = tt * cc;
                for (int k = 0; k < 3; ++k) {
                    float kp = K[k][pp], kq = K[k][qq];
                    K[k][pp] = cc * kp - ss * kq;
                    K[k][qq] = ss * kp + cc * kq;
                }
                for (int k = 0; k < 3; ++k) {
                    float kp = K[pp][k], kq = K[qq][k];
                    K[pp][k] = cc * kp - ss * kq;
                    K[qq][k] = ss * kp + cc * kq;
                }
                for (int k = 0; k < 3; ++k) {
                    float vp = V[k][pp], vq = V[k][qq];
                    V[k][pp] = cc * vp - ss * vq;
                    V[k][qq] = ss * vp + cc * vq;
                }
            }
        }
        float eig[3] = { K[0][0], K[1][1], K[2][2] };
        // sort descending (so reflection fix hits the smallest singular value)
        for (int i = 0; i < 2; ++i)
            for (int j = 0; j < 2 - i; ++j)
                if (eig[j] < eig[j + 1]) {
                    float te = eig[j]; eig[j] = eig[j + 1]; eig[j + 1] = te;
                    for (int k = 0; k < 3; ++k) {
                        float tv = V[k][j]; V[k][j] = V[k][j + 1]; V[k][j + 1] = tv;
                    }
                }
        // U columns: u_i = H v_i / s_i
        float U[3][3];
        for (int i = 0; i < 3; ++i) {
            float u0 = H[0][0] * V[0][i] + H[0][1] * V[1][i] + H[0][2] * V[2][i];
            float u1 = H[1][0] * V[0][i] + H[1][1] * V[1][i] + H[1][2] * V[2][i];
            float u2 = H[2][0] * V[0][i] + H[2][1] * V[1][i] + H[2][2] * V[2][i];
            float nrm = sqrtf(u0 * u0 + u1 * u1 + u2 * u2);
            if (nrm > 1e-20f) {
                float r = 1.0f / nrm;
                U[0][i] = u0 * r; U[1][i] = u1 * r; U[2][i] = u2 * r;
            } else {
                // degenerate: complete with cross product of the other columns
                int a = (i + 1) % 3, c = (i + 2) % 3;
                U[0][i] = U[1][a] * U[2][c] - U[2][a] * U[1][c];
                U[1][i] = U[2][a] * U[0][c] - U[0][a] * U[2][c];
                U[2][i] = U[0][a] * U[1][c] - U[1][a] * U[0][c];
            }
        }
        // R0 = V U^T, det, reflection fix on V's last column
        float R0[3][3];
        for (int d = 0; d < 3; ++d)
            for (int e = 0; e < 3; ++e)
                R0[d][e] = V[d][0] * U[e][0] + V[d][1] * U[e][1] + V[d][2] * U[e][2];
        float det = R0[0][0] * (R0[1][1] * R0[2][2] - R0[1][2] * R0[2][1])
                  - R0[0][1] * (R0[1][0] * R0[2][2] - R0[1][2] * R0[2][0])
                  + R0[0][2] * (R0[1][0] * R0[2][1] - R0[1][1] * R0[2][0]);
        float flip = (det < 0.0f) ? -1.0f : 1.0f;
        float R[3][3];
        for (int d = 0; d < 3; ++d)
            for (int e = 0; e < 3; ++e)
                R[d][e] = V[d][0] * U[e][0] + V[d][1] * U[e][1] + flip * V[d][2] * U[e][2];

        float* Ro = out + (size_t)b * 9;
        for (int d = 0; d < 3; ++d)
            for (int e = 0; e < 3; ++e)
                Ro[d * 3 + e] = R[d][e];
        float* to = out + (size_t)BATCH * 9 + (size_t)b * 3;
        for (int d = 0; d < 3; ++d)
            to[d] = -(R[d][0] * mu[0] + R[d][1] * mu[1] + R[d][2] * mu[2]) + nu[d];
    }
}

extern "C" void kernel_launch(void* const* d_in, const int* in_sizes, int n_in,
                              void* d_out, int out_size, void* d_ws, size_t ws_size,
                              hipStream_t stream) {
    const float* src    = (const float*)d_in[0];   // [64,3,1024]
    const float* tgt    = (const float*)d_in[1];   // [64,3,1024]
    const float* scores = (const float*)d_in[2];   // [64,1024,1024]
    float* out  = (float*)d_out;                   // R[64,3,3] ++ t[64,3]
    float* corr = (float*)d_ws;                    // [64,3,1024] scratch (768 KB)

    svdhead_corr_kernel<<<dim3(BATCH * 8), dim3(256), 0, stream>>>(tgt, scores, corr);
    svdhead_kabsch_kernel<<<dim3(BATCH), dim3(256), 0, stream>>>(src, corr, out);
}
